// Model_89953795048037
// MI455X (gfx1250) — matrix operand order repaired and checked by reading, not compiled, not run on hardware
//
#include <hip/hip_runtime.h>
#include <hip/hip_bf16.h>

// Problem constants
constexpr int Bc   = 64;
constexpr int Tc   = 1000;
constexpr int Ic   = 129;
constexpr int IPAD = 160;    // K padded to multiple of 32 for WMMA
constexpr int Hc   = 512;
constexpr int Gc   = 2048;   // 4*H
constexpr int Cc   = 129;
constexpr int CPAD = 144;    // N padded to multiple of 16
constexpr long Mc  = (long)Bc * Tc;   // 64000 rows

// LDS stride (halves) for the h buffer: 520*2B = 1040B, 16B aligned rows
constexpr int HSTR = 520;

typedef __attribute__((ext_vector_type(16))) _Float16 v16h;
typedef __attribute__((ext_vector_type(8)))  _Float16 v8h;
typedef __attribute__((ext_vector_type(8)))  float    v8f;

union V16U { v16h v; v8h h[2]; };

// ---- WMMA operand loaders (wave32, 16x16x32 f16) ------------------------
// A operand (16x32, M x K): lane L holds row (L&15);
//   lanes 0-15: K {0..7} and {16..23}; lanes 16-31: K {8..15} and {24..31}
// rowp = &src[row * stride + kt*32]
__device__ __forceinline__ v16h load_opA(const _Float16* rowp, int lane) {
  const _Float16* p = rowp + ((lane & 16) ? 8 : 0);
  V16U u;
  u.h[0] = *(const v8h*)(p);
  u.h[1] = *(const v8h*)(p + 16);
  return u.v;
}
// B operand (32x16, K x N): lane L holds column (L&15);
//   lanes 0-15: K {0..7} and {16..23}; lanes 16-31: K {8..15} and {24..31}
// For D = A * W^T the B operand is just row n of row-major W.
// rowp = &W[n * stride + kt*32]
__device__ __forceinline__ v16h load_opB(const _Float16* rowp, int lane) {
  const _Float16* p = rowp + ((lane & 16) ? 8 : 0);
  V16U u;
  u.h[0] = *(const v8h*)(p);
  u.h[1] = *(const v8h*)(p + 16);
  return u.v;
}

__device__ __forceinline__ float sigm(float x) {
  return 1.f / (1.f + __expf(-x));
}

// ---- Kernel 0a: f32 -> f16 convert with zero padding --------------------
__global__ void k_cvt_pad(const float* __restrict__ src, _Float16* __restrict__ dst,
                          int Rin, int Cin, int Rout, int Cout) {
  long n = (long)Rout * Cout;
  long stride = (long)gridDim.x * blockDim.x;
  for (long i = (long)blockIdx.x * blockDim.x + threadIdx.x; i < n; i += stride) {
    int r = (int)(i / Cout);
    int c = (int)(i % Cout);
    float v = (r < Rin && c < Cin) ? src[(long)r * Cin + c] : 0.f;
    dst[i] = (_Float16)v;
  }
}

// ---- Kernel 0b: W_hh convert + gate-interleave row permutation ----------
// Permuted row n' = w*64 + gate*16 + r   <-   original row gate*512 + w*16 + r
// so that wave w's 4 N-tiles are gates {i,f,g,o} for h-columns [16w,16w+16).
__global__ void k_cvt_whh(const float* __restrict__ src, _Float16* __restrict__ dst) {
  long n = (long)Gc * Hc;
  long stride = (long)gridDim.x * blockDim.x;
  for (long i = (long)blockIdx.x * blockDim.x + threadIdx.x; i < n; i += stride) {
    int np = (int)(i / Hc);
    int k  = (int)(i % Hc);
    int gate = (np >> 4) & 3;
    int w    = np >> 6;
    int rr   = np & 15;
    int orig = gate * Hc + w * 16 + rr;
    dst[i] = (_Float16)src[(long)orig * Hc + k];
  }
}

// ---- Kernel 1: x_proj = X * W_ih^T + (b_ih + b_hh), stored f16 ----------
// grid = M/16 blocks, 512 threads (16 waves); wave handles 8 N-tiles.
__global__ __launch_bounds__(512)
void k_xproj(const _Float16* __restrict__ inp,   // [M][IPAD]
             const _Float16* __restrict__ wih,   // [G][IPAD]
             const float* __restrict__ bih, const float* __restrict__ bhh,
             _Float16* __restrict__ xp) {        // [M][G]
  int lane = threadIdx.x & 31;
  int wave = threadIdx.x >> 5;        // 0..15
  int r    = lane & 15;
  int mh   = (lane & 16) ? 8 : 0;
  long m0  = (long)blockIdx.x * 16;
  const _Float16* arow = inp + (m0 + r) * IPAD;

  v8f acc[8] = {};
  for (int kt = 0; kt < IPAD / 32; ++kt) {
    v16h a = load_opA(arow + kt * 32, lane);
#pragma unroll
    for (int nt = 0; nt < 8; ++nt) {
      int n = (wave * 8 + nt) * 16 + r;
      v16h bm = load_opB(wih + (long)n * IPAD + kt * 32, lane);
      acc[nt] = __builtin_amdgcn_wmma_f32_16x16x32_f16(false, a, false, bm,
                                                       (short)0, acc[nt], false, false);
    }
  }
#pragma unroll
  for (int nt = 0; nt < 8; ++nt) {
    int col = (wave * 8 + nt) * 16 + r;
    float bias = bih[col] + bhh[col];
#pragma unroll
    for (int g = 0; g < 8; ++g) {
      long row = m0 + g + mh;
      xp[row * Gc + col] = (_Float16)(acc[nt][g] + bias);
    }
  }
}

// ---- Kernel 2: the LSTM recurrence --------------------------------------
// 4 workgroups (16 batch rows each), 1024 threads = 32 waves.
// W_hh is gate-permuted, so wave w's 4 accumulators are exactly the i,f,g,o
// gates of h-columns [16w,16w+16): the entire LSTM cell update happens on
// register-resident WMMA accumulators (c stays in registers for all T steps).
// h is double-buffered in LDS -> a single barrier per timestep.
__global__ __launch_bounds__(1024)
void k_lstm(const _Float16* __restrict__ whh_p, // permuted [G][H]
            const _Float16* __restrict__ xp,    // [B*T][G] f16 (original gate order)
            _Float16* __restrict__ hs) {        // [B*T][H] f16
  __shared__ alignas(16) _Float16 hbuf[2][16 * HSTR];

  int tid  = threadIdx.x;
  int lane = tid & 31;
  int wave = tid >> 5;               // 0..31
  int r    = lane & 15;
  int mh   = (lane & 16) ? 8 : 0;
  int b0   = blockIdx.x * 16;        // first batch row of this workgroup
  int jcol = wave * 16 + r;          // h column owned by this lane

  for (int i = tid; i < 16 * HSTR; i += 1024) hbuf[0][i] = (_Float16)0.f;
  float c_reg[8];
#pragma unroll
  for (int g = 0; g < 8; ++g) c_reg[g] = 0.f;
  __syncthreads();

  for (int t = 0; t < Tc; ++t) {
    const _Float16* hcur = hbuf[t & 1];
    _Float16*       hnxt = hbuf[(t + 1) & 1];

    // --- WMMA: i,f,g,o gate tiles for this wave's 16 h-columns ---
    v8f acc[4] = {};
    for (int kt = 0; kt < Hc / 32; ++kt) {
      v16h a = load_opA(hcur + r * HSTR + kt * 32, lane);
#pragma unroll
      for (int nt = 0; nt < 4; ++nt) {
        int n = wave * 64 + nt * 16 + r;     // permuted weight row
        v16h bm = load_opB(whh_p + (long)n * Hc + kt * 32, lane);
        acc[nt] = __builtin_amdgcn_wmma_f32_16x16x32_f16(false, a, false, bm,
                                                         (short)0, acc[nt], false, false);
      }
    }

    // --- register-resident LSTM cell update (8 batch rows per lane) ---
#pragma unroll
    for (int g = 0; g < 8; ++g) {
      int m = g + mh;
      const _Float16* xr = xp + ((long)(b0 + m) * Tc + t) * Gc;
      __builtin_prefetch(xr + Gc, 0, 1);     // next timestep slice
      float iv = sigm (acc[0][g] + (float)xr[jcol]);
      float fv = sigm (acc[1][g] + (float)xr[jcol + 512]);
      float gv = tanhf(acc[2][g] + (float)xr[jcol + 1024]);
      float ov = sigm (acc[3][g] + (float)xr[jcol + 1536]);
      float cn = fv * c_reg[g] + iv * gv;
      c_reg[g] = cn;
      float hn = ov * tanhf(cn);
      hnxt[m * HSTR + jcol] = (_Float16)hn;
      hs[((long)(b0 + m) * Tc + t) * Hc + jcol] = (_Float16)hn;
    }
    __syncthreads();   // hnxt complete; safe because hcur/hnxt alternate
  }
}

// ---- Kernel 3: logits = hs @ W_lin^T + b_lin ----------------------------
// grid = M/16, 288 threads (9 waves); wave nt covers output cols nt*16..+15.
__global__ __launch_bounds__(288)
void k_logits(const _Float16* __restrict__ hs,    // [M][H]
              const _Float16* __restrict__ wlin,  // [CPAD][H] zero-padded rows
              const float* __restrict__ blin,
              float* __restrict__ out) {          // [M][C]
  int lane = threadIdx.x & 31;
  int wave = threadIdx.x >> 5;        // 0..8
  int r    = lane & 15;
  int mh   = (lane & 16) ? 8 : 0;
  long m0  = (long)blockIdx.x * 16;
  const _Float16* arow = hs + (m0 + r) * Hc;
  int n = wave * 16 + r;

  v8f acc = {};
  for (int kt = 0; kt < Hc / 32; ++kt) {
    v16h a  = load_opA(arow + kt * 32, lane);
    v16h bm = load_opB(wlin + (long)n * Hc + kt * 32, lane);
    acc = __builtin_amdgcn_wmma_f32_16x16x32_f16(false, a, false, bm,
                                                 (short)0, acc, false, false);
  }
  if (n < Cc) {
    float bias = blin[n];
#pragma unroll
    for (int g = 0; g < 8; ++g) {
      long row = m0 + g + mh;
      out[row * Cc + n] = acc[g] + bias;
    }
  }
}

// -------------------------------------------------------------------------
extern "C" void kernel_launch(void* const* d_in, const int* in_sizes, int n_in,
                              void* d_out, int out_size, void* d_ws, size_t ws_size,
                              hipStream_t stream) {
  const float* inputs = (const float*)d_in[0];  // [B,T,I]
  const float* Wih    = (const float*)d_in[1];  // [4H,I]
  const float* Whh    = (const float*)d_in[2];  // [4H,H]
  const float* bih    = (const float*)d_in[3];  // [4H]
  const float* bhh    = (const float*)d_in[4];  // [4H]
  const float* Wlin   = (const float*)d_in[5];  // [C,H]
  const float* blin   = (const float*)d_in[6];  // [C]
  float* out = (float*)d_out;                   // [B,T,C]

  char* ws = (char*)d_ws;
  size_t off = 0;
  auto salloc = [&](size_t bytes) {
    char* p = ws + off;
    off = (off + bytes + 255) & ~(size_t)255;
    return p;
  };
  _Float16* inp16  = (_Float16*)salloc((size_t)Mc * IPAD * 2);
  _Float16* wih16  = (_Float16*)salloc((size_t)Gc * IPAD * 2);
  _Float16* whh16  = (_Float16*)salloc((size_t)Gc * Hc * 2);   // gate-permuted
  _Float16* wlin16 = (_Float16*)salloc((size_t)CPAD * Hc * 2);
  _Float16* xp16   = (_Float16*)salloc((size_t)Mc * Gc * 2);
  _Float16* hs16   = (_Float16*)salloc((size_t)Mc * Hc * 2);

  // Converts / padding / gate permutation
  k_cvt_pad<<<1024, 256, 0, stream>>>(inputs, inp16, (int)Mc, Ic, (int)Mc, IPAD);
  k_cvt_pad<<<256,  256, 0, stream>>>(Wih,  wih16,  Gc, Ic, Gc, IPAD);
  k_cvt_whh<<<512,  256, 0, stream>>>(Whh,  whh16);
  k_cvt_pad<<<64,   256, 0, stream>>>(Wlin, wlin16, Cc, Hc, CPAD, Hc);

  // Big parallel GEMM: input projections for all timesteps
  k_xproj<<<(int)(Mc / 16), 512, 0, stream>>>(inp16, wih16, bih, bhh, xp16);

  // Serial recurrence: 4 independent batch groups, one barrier per step
  k_lstm<<<4, 1024, 0, stream>>>(whh16, xp16, hs16);

  // Big parallel GEMM: per-step classifier
  k_logits<<<(int)(Mc / 16), 288, 0, stream>>>(hs16, wlin16, blin, out);
}
